// NSFPosteriorFlow_17652315587311
// MI455X (gfx1250) — matrix operand order repaired and checked by reading, not compiled, not run on hardware
//
#include <hip/hip_runtime.h>
#include <hip/hip_bf16.h>

typedef __attribute__((ext_vector_type(16))) _Float16 v16h;
typedef __attribute__((ext_vector_type(8)))  _Float16 v8h;
typedef __attribute__((ext_vector_type(8)))  float    v8f;

#define NSF_F     15
#define NSF_H     128
#define NSF_L     12
#define NSF_NB    16
#define NSF_M     47      // 3*NB - 1
#define NSF_TAIL  3.0f
#define NSF_MINW  0.001f
#define NSF_MINH  0.001f
#define NSF_MIND  0.001f

// fragment layout: per layer 316 fragments of 512 halves each
//   [0,8)        init  : ntile(8) x kstep(1)          B = (W_init*m_init)^T  K=32pad x N=128
//   [8,136)      res   : mat(4) x ntile(8) x kstep(4) B = (W_res*m_h)^T      K=128 x N=128
//   [136,316)    out   : ntile(45) x kstep(4)         B = (W_out*m_out)^T    K=128 x N=720 (48/feature, col 47 zero)
#define FRAGS_PER_LAYER 316
#define FRAG_HALVES     512

__device__ __forceinline__ int deg_h(int n) { return (n % (NSF_F - 1)) + 1; }

// ---------------------------------------------------------------------------
// Weight packing kernel: one block (32 lanes) per fragment.
// B fragment layout (dense 16-bit B, 32xK16xN, wave32):
//   lane L: N = L%16 ; half i (0..15): K = kstep*32 + 16*(L/16) + i
// ---------------------------------------------------------------------------
__global__ void nsf_pack_weights(const float* __restrict__ Wi,
                                 const float* __restrict__ Wr,
                                 const float* __restrict__ Wo,
                                 _Float16* __restrict__ ws) {
  const int frag = blockIdx.x;
  const int lane = threadIdx.x;
  const int l  = frag / FRAGS_PER_LAYER;
  const int id = frag % FRAGS_PER_LAYER;
  const int nn  = lane & 15;
  const int khi = (lane >> 4) * 16;

  v16h out;
  if (id < 8) {                      // init matmul weights
    const int n  = id * 16 + nn;
    const int dh = deg_h(n);
#pragma unroll
    for (int i = 0; i < 16; ++i) {
      const int K = khi + i;
      float v = 0.f;
      if (K < NSF_F && dh >= K + 1) v = Wi[((size_t)l * NSF_H + n) * NSF_F + K];
      out[i] = (_Float16)v;
    }
  } else if (id < 136) {             // residual matmul weights
    const int r = id - 8;
    const int m = r >> 5;            // blk*2 + sub
    const int ntile = (r & 31) >> 2;
    const int kk = r & 3;
    const int n  = ntile * 16 + nn;
    const int dh = deg_h(n);
#pragma unroll
    for (int i = 0; i < 16; ++i) {
      const int K = kk * 32 + khi + i;
      float v = 0.f;
      if (dh >= deg_h(K)) v = Wr[(((size_t)l * 4 + m) * NSF_H + n) * NSF_H + K];
      out[i] = (_Float16)v;
    }
  } else {                           // output head weights (47 -> 48 padded cols)
    const int r = id - 136;
    const int ntile = r >> 2;
    const int kk = r & 3;
    const int oc = ntile * 16 + nn;  // padded column, 0..719
    const int f  = oc / 48;
    const int j  = oc % 48;
#pragma unroll
    for (int i = 0; i < 16; ++i) {
      const int K = kk * 32 + khi + i;
      float v = 0.f;
      if (j < NSF_M && (f + 1) > deg_h(K))
        v = Wo[((size_t)l * (NSF_F * NSF_M) + f * NSF_M + j) * NSF_H + K];
      out[i] = (_Float16)v;
    }
  }
  ((v16h*)ws)[(size_t)frag * 32 + lane] = out;
}

// ---------------------------------------------------------------------------
// Main fused flow kernel
// ---------------------------------------------------------------------------
struct alignas(16) WaveShared {
  float    xcur[16][16];      // carry (post-transform features)
  float    xrevf[16][16];     // reversed features f32 (this layer's x)
  _Float16 xrevh[16][32];     // reversed features f16, K padded to 32
  float    hbuf[16][NSF_H];   // hidden state f32
  _Float16 a16[16][NSF_H];    // f16 A-source (relu(h) / h)
  _Float16 t16[16][NSF_H];    // f16 A-source (relu(t1))
  float    outbuf[16][48];    // one feature's spline params (padded 48)
};

__device__ __forceinline__ void wsync() {
  __builtin_amdgcn_wave_barrier();
#if __has_builtin(__builtin_amdgcn_s_wait_dscnt)
  __builtin_amdgcn_s_wait_dscnt(0);
  asm volatile("" ::: "memory");
#else
  asm volatile("s_wait_dscnt 0" ::: "memory");
#endif
  __builtin_amdgcn_wave_barrier();
}

// Build A fragment (16x32 f16, M=rows) from row-major f16 LDS buffer.
// lane L: M = L%16; half i (0..15) -> K = kk*32 + 16*(L/16) + i
__device__ __forceinline__ v16h load_a_frag(const _Float16* base, int pitch,
                                            int kk, int lane) {
  const int m  = lane & 15;
  const int hi = lane >> 4;
  const _Float16* p = base + (size_t)m * pitch + kk * 32 + hi * 16;
  v8h lo = *(const v8h*)(p);
  v8h hh = *(const v8h*)(p + 8);
  v16h r;
#pragma unroll
  for (int i = 0; i < 8; ++i) { r[i] = lo[i]; r[8 + i] = hh[i]; }
  return r;
}

__device__ __forceinline__ float softplusf(float x) {
  return (x > 20.f) ? x : log1pf(expf(x));
}

__global__ __launch_bounds__(256, 1) void nsf_main(
    const float* __restrict__ x,
    const float* __restrict__ b_init,
    const float* __restrict__ b_res,
    const float* __restrict__ b_out,
    const _Float16* __restrict__ wpack,
    float* __restrict__ z_out,
    float* __restrict__ ld_out) {
  extern __shared__ unsigned char smem_raw[];
  const int lane = threadIdx.x & 31;
  const int wav  = threadIdx.x >> 5;
  WaveShared* S  = ((WaveShared*)smem_raw) + wav;
  const int rowbase = blockIdx.x * 128 + wav * 16;
  const v16h* WF = (const v16h*)wpack;
  const int nn = lane & 15;
  const int hi = lane >> 4;

  // load 16 rows of x
  for (int t = lane; t < 16 * NSF_F; t += 32) {
    const int r = t / NSF_F, f = t % NSF_F;
    S->xcur[r][f] = x[(size_t)(rowbase + r) * NSF_F + f];
  }
  float ld = 0.f;
  wsync();

  for (int l = 0; l < NSF_L; ++l) {
    const size_t lb = (size_t)l * FRAGS_PER_LAYER;

    // --- reverse features: xrev[f] = xcur[14-f] ---
    for (int t = lane; t < 256; t += 32) {
      const int r = t >> 4, f = t & 15;
      S->xrevf[r][f] = (f < NSF_F) ? S->xcur[r][NSF_F - 1 - f] : 0.f;
    }
    wsync();
    for (int t = lane; t < 512; t += 32) {
      const int r = t >> 5, k = t & 31;
      S->xrevh[r][k] = (k < NSF_F) ? (_Float16)S->xrevf[r][k] : (_Float16)0.f;
    }
    wsync();

    // --- init matmul: h = xrev @ (W_init*m)^T + b ---
    {
      v16h a = load_a_frag(&S->xrevh[0][0], 32, 0, lane);
#pragma unroll
      for (int nt = 0; nt < 8; ++nt) {
        v16h b = WF[(lb + nt) * 32 + lane];
        v8f c = {};
        c = __builtin_amdgcn_wmma_f32_16x16x32_f16(false, a, false, b,
                                                   (short)0, c, false, false);
        const int n = nt * 16 + nn;
        const float bias = b_init[l * NSF_H + n];
#pragma unroll
        for (int r2 = 0; r2 < 8; ++r2) {
          const int row = hi * 8 + r2;
          const float v = c[r2] + bias;
          S->hbuf[row][n] = v;
          S->a16[row][n]  = (_Float16)fmaxf(v, 0.f);
        }
      }
    }
    wsync();

    // --- 2 residual blocks: t1=relu(h)W0+b0; t2=relu(t1)W1+b1; h+=t2 ---
    for (int blk = 0; blk < 2; ++blk) {
      for (int sub = 0; sub < 2; ++sub) {
        const _Float16* srcA = (sub == 0) ? &S->a16[0][0] : &S->t16[0][0];
        const int m = blk * 2 + sub;
        for (int nt = 0; nt < 8; ++nt) {
          v8f c = {};
#pragma unroll
          for (int kk = 0; kk < 4; ++kk) {
            v16h a = load_a_frag(srcA, NSF_H, kk, lane);
            v16h b = WF[(lb + 8 + m * 32 + nt * 4 + kk) * 32 + lane];
            c = __builtin_amdgcn_wmma_f32_16x16x32_f16(false, a, false, b,
                                                       (short)0, c, false, false);
          }
          const int n = nt * 16 + nn;
          const float bias = b_res[((size_t)l * 4 + m) * NSF_H + n];
#pragma unroll
          for (int r2 = 0; r2 < 8; ++r2) {
            const int row = hi * 8 + r2;
            const float v = c[r2] + bias;
            if (sub == 0) {
              S->t16[row][n] = (_Float16)fmaxf(v, 0.f);
            } else {
              const float nh = S->hbuf[row][n] + v;
              S->hbuf[row][n] = nh;
              // after blk0: relu(h) feeds blk1; after blk1: plain h feeds out head
              S->a16[row][n] = (_Float16)((blk == 1) ? nh : fmaxf(nh, 0.f));
            }
          }
        }
        wsync();
      }
    }

    // --- output head + RQS, one feature at a time (3 N-tiles of 16) ---
    for (int f = 0; f < NSF_F; ++f) {
      for (int t3 = 0; t3 < 3; ++t3) {
        const int nt = f * 3 + t3;
        v8f c = {};
#pragma unroll
        for (int kk = 0; kk < 4; ++kk) {
          v16h a = load_a_frag(&S->a16[0][0], NSF_H, kk, lane);
          v16h b = WF[(lb + 136 + nt * 4 + kk) * 32 + lane];
          c = __builtin_amdgcn_wmma_f32_16x16x32_f16(false, a, false, b,
                                                     (short)0, c, false, false);
        }
        const int j = t3 * 16 + nn;           // 0..47 within feature
        const float bias = (j < NSF_M) ? b_out[(size_t)l * (NSF_F * NSF_M) + f * NSF_M + j] : 0.f;
#pragma unroll
        for (int r2 = 0; r2 < 8; ++r2)
          S->outbuf[hi * 8 + r2][j] = c[r2] + bias;
      }
      wsync();

      if (lane < 16) {  // one lane per row: rational quadratic spline
        const int row = lane;
        const float xval = S->xrevf[row][f];
        const bool inside = (xval >= -NSF_TAIL) && (xval <= NSF_TAIL);
        const float xc = fminf(fmaxf(xval, -NSF_TAIL), NSF_TAIL);
        const float* P = &S->outbuf[row][0];

        // bin widths from softmax(uw)
        float mw = P[0];
#pragma unroll
        for (int i = 1; i < NSF_NB; ++i) mw = fmaxf(mw, P[i]);
        float ew[NSF_NB]; float se = 0.f;
#pragma unroll
        for (int i = 0; i < NSF_NB; ++i) { ew[i] = expf(P[i] - mw); se += ew[i]; }
        float cw[NSF_NB + 1]; cw[0] = -NSF_TAIL;
        {
          const float scl = (1.f - NSF_MINW * NSF_NB) / se;
          float cum = 0.f;
#pragma unroll
          for (int i = 0; i < NSF_NB; ++i) {
            cum += NSF_MINW + scl * ew[i];
            cw[i + 1] = (i == NSF_NB - 1) ? NSF_TAIL : 2.f * NSF_TAIL * cum - NSF_TAIL;
          }
        }
        // bin heights from softmax(uh)
        float mh = P[NSF_NB];
#pragma unroll
        for (int i = 1; i < NSF_NB; ++i) mh = fmaxf(mh, P[NSF_NB + i]);
        float eh[NSF_NB]; float sh = 0.f;
#pragma unroll
        for (int i = 0; i < NSF_NB; ++i) { eh[i] = expf(P[NSF_NB + i] - mh); sh += eh[i]; }
        float ch[NSF_NB + 1]; ch[0] = -NSF_TAIL;
        {
          const float scl = (1.f - NSF_MINH * NSF_NB) / sh;
          float cum = 0.f;
#pragma unroll
          for (int i = 0; i < NSF_NB; ++i) {
            cum += NSF_MINH + scl * eh[i];
            ch[i + 1] = (i == NSF_NB - 1) ? NSF_TAIL : 2.f * NSF_TAIL * cum - NSF_TAIL;
          }
        }
        // derivatives (padded with 1.0 at both ends)
        float dv[NSF_NB + 1];
        dv[0] = 1.f; dv[NSF_NB] = 1.f;
#pragma unroll
        for (int i = 1; i < NSF_NB; ++i) dv[i] = NSF_MIND + softplusf(P[2 * NSF_NB + i - 1]);

        // select bin (static-index selection over unrolled loop)
        float in_cw = cw[0], in_w = cw[1] - cw[0];
        float in_ch = ch[0], in_h = ch[1] - ch[0];
        float in_d = dv[0], in_d1 = dv[1];
#pragma unroll
        for (int i = 1; i < NSF_NB; ++i) {
          if (xc >= cw[i]) {
            in_cw = cw[i]; in_w = cw[i + 1] - cw[i];
            in_ch = ch[i]; in_h = ch[i + 1] - ch[i];
            in_d = dv[i];  in_d1 = dv[i + 1];
          }
        }
        const float delta = in_h / in_w;
        const float th   = (xc - in_cw) / in_w;
        const float om   = 1.f - th;
        const float tomt = th * om;
        const float num  = in_h * (delta * th * th + in_d * tomt);
        const float den  = delta + (in_d + in_d1 - 2.f * delta) * tomt;
        float y = in_ch + num / den;
        const float dnum = delta * delta *
            (in_d1 * th * th + 2.f * delta * tomt + in_d * om * om);
        float lad = logf(dnum) - 2.f * logf(den);
        y   = inside ? y : xval;
        lad = inside ? lad : 0.f;
        S->xcur[row][f] = y;   // new carry (reversed-order indexing)
        ld += lad;
      }
      wsync();
    }
  }

  // write z and logdet (d_out = [z (B,15) | logdet (B)])
  for (int t = lane; t < 16 * NSF_F; t += 32) {
    const int r = t / NSF_F, f = t % NSF_F;
    z_out[(size_t)(rowbase + r) * NSF_F + f] = S->xcur[r][f];
  }
  if (lane < 16) ld_out[rowbase + lane] = ld;
}

// ---------------------------------------------------------------------------
extern "C" void kernel_launch(void* const* d_in, const int* in_sizes, int n_in,
                              void* d_out, int out_size, void* d_ws, size_t ws_size,
                              hipStream_t stream) {
  const float* x      = (const float*)d_in[0];
  const float* W_init = (const float*)d_in[1];
  const float* b_init = (const float*)d_in[2];
  const float* W_res  = (const float*)d_in[3];
  const float* b_res  = (const float*)d_in[4];
  const float* W_out  = (const float*)d_in[5];
  const float* b_out  = (const float*)d_in[6];

  const int B = in_sizes[0] / NSF_F;          // 65536
  _Float16* wpack = (_Float16*)d_ws;

  // pack masked weights into WMMA B-fragment layout (f16)
  const int nfrag = NSF_L * FRAGS_PER_LAYER;  // 3792
  nsf_pack_weights<<<nfrag, 32, 0, stream>>>(W_init, W_res, W_out, wpack);

  float* z   = (float*)d_out;
  float* ldt = z + (size_t)B * NSF_F;

  const size_t shmem = 8 * sizeof(WaveShared);  // 8 waves * 22.5KB = 180KB
  static_assert(sizeof(WaveShared) == 22528, "LDS layout size");
  (void)hipFuncSetAttribute((const void*)nsf_main,
                            hipFuncAttributeMaxDynamicSharedMemorySize,
                            (int)shmem);

  const int rows_per_block = 128;               // 8 waves * 16 rows
  nsf_main<<<B / rows_per_block, 256, shmem, stream>>>(
      x, b_init, b_res, b_out, wpack, z, ldt);
}
